// Bert_BiLSTM_CRF_876173328468
// MI455X (gfx1250) — matrix operand order repaired and checked by reading, not compiled, not run on hardware
//
#include <hip/hip_runtime.h>
#include <hip/hip_bf16.h>
#include <math.h>

// ---------------------------------------------------------------------------
// Bert_BiLSTM_CRF for MI455X (gfx1250): WMMA bf16 GEMMs + VALU tail kernels.
// ---------------------------------------------------------------------------

typedef __attribute__((ext_vector_type(16))) __bf16 v16bf;
typedef __attribute__((ext_vector_type(8)))  float  v8f;

#define DMODEL 768
#define NHEAD  12
#define DHEAD  64
#define DFFN   3072
#define BSZ    8
#define SL     256
#define MROWS  (BSZ * SL)        // 2048
#define HID    384
#define G4     (4 * HID)         // 1536
#define NTAG   9
#define TAG_CLS 7
#define TAG_SEP 8
#define NEGINF (-10000.0f)
#define NLAY   12

__device__ __forceinline__ unsigned short f2bfu(float f) {
  unsigned u = __builtin_bit_cast(unsigned, f);
  u += 0x7FFFu + ((u >> 16) & 1u);            // round-to-nearest-even
  return (unsigned short)(u >> 16);
}
__device__ __forceinline__ unsigned pk2bf(float lo, float hi) {
  return (unsigned)f2bfu(lo) | ((unsigned)f2bfu(hi) << 16);
}
__device__ __forceinline__ float sigmoidf_(float x) { return 1.0f / (1.0f + __expf(-x)); }

// ---------------------------------------------------------------------------
// Embedding gather + fused LayerNorm.  One block (256 thr) per token row.
// ---------------------------------------------------------------------------
__global__ __launch_bounds__(256)
void embed_ln_kernel(const int* __restrict__ ids, const float* __restrict__ we,
                     const float* __restrict__ pe, const float* __restrict__ te,
                     const float* __restrict__ g,  const float* __restrict__ be,
                     float* __restrict__ out) {
  const int r = blockIdx.x;            // 0..2047
  const int l = r % SL;
  const int id = ids[r];
  __shared__ float rowv[DMODEL];
  __shared__ float red[256];
  float s1 = 0.f, s2 = 0.f;
  for (int d = threadIdx.x; d < DMODEL; d += 256) {
    float v = we[(size_t)id * DMODEL + d] + pe[(size_t)l * DMODEL + d] + te[d];
    rowv[d] = v; s1 += v; s2 += v * v;
  }
  red[threadIdx.x] = s1; __syncthreads();
  for (int s = 128; s > 0; s >>= 1) { if (threadIdx.x < s) red[threadIdx.x] += red[threadIdx.x + s]; __syncthreads(); }
  const float mean = red[0] / (float)DMODEL; __syncthreads();
  red[threadIdx.x] = s2; __syncthreads();
  for (int s = 128; s > 0; s >>= 1) { if (threadIdx.x < s) red[threadIdx.x] += red[threadIdx.x + s]; __syncthreads(); }
  const float var = red[0] / (float)DMODEL - mean * mean;
  const float inv = rsqrtf(var + 1e-12f);
  for (int d = threadIdx.x; d < DMODEL; d += 256)
    out[(size_t)r * DMODEL + d] = (rowv[d] - mean) * inv * g[d] + be[d];
}

// ---------------------------------------------------------------------------
// Row LayerNorm (separate in/out).  One block per row.
// ---------------------------------------------------------------------------
__global__ __launch_bounds__(256)
void layernorm_kernel(const float* __restrict__ x, const float* __restrict__ g,
                      const float* __restrict__ be, float* __restrict__ out, int ncol) {
  const int r = blockIdx.x;
  __shared__ float red[256];
  float s1 = 0.f, s2 = 0.f;
  for (int d = threadIdx.x; d < ncol; d += 256) {
    float v = x[(size_t)r * ncol + d]; s1 += v; s2 += v * v;
  }
  red[threadIdx.x] = s1; __syncthreads();
  for (int s = 128; s > 0; s >>= 1) { if (threadIdx.x < s) red[threadIdx.x] += red[threadIdx.x + s]; __syncthreads(); }
  const float mean = red[0] / (float)ncol; __syncthreads();
  red[threadIdx.x] = s2; __syncthreads();
  for (int s = 128; s > 0; s >>= 1) { if (threadIdx.x < s) red[threadIdx.x] += red[threadIdx.x + s]; __syncthreads(); }
  const float var = red[0] / (float)ncol - mean * mean;
  const float inv = rsqrtf(var + 1e-12f);
  for (int d = threadIdx.x; d < ncol; d += 256)
    out[(size_t)r * ncol + d] = (x[(size_t)r * ncol + d] - mean) * inv * g[d] + be[d];
}

// ---------------------------------------------------------------------------
// Tiled GEMM:  C[M,N] = A[M,K] @ B(+epilogue).   bf16 WMMA, f32 accumulate.
// PRECONDITIONS: M % 128 == 0, N % 128 == 0, K % 32 == 0  (all call sites do).
//   transB=0 : B row-major [K,N];  transB=1 : B row-major [N,K] (use B^T)
//   mode: 0 none | 1 +bias | 2 +bias,GELU(erf) | 3 +bias,+Res
// Block 256 thr (8 waves); 128x128 C tile; BK=32; double-buffered LDS tiles
// stored in WMMA fragment order (each fragment = one 32B contiguous record),
// staged with b128 global loads and packed b64/b128 ds stores (branch-free).
// ---------------------------------------------------------------------------
__global__ __launch_bounds__(256)
void gemm_bf16_wmma(const float* __restrict__ A, const float* __restrict__ Bm,
                    const float* __restrict__ bias, const float* __restrict__ Res,
                    float* __restrict__ C, int M, int N, int K, int transB, int mode) {
  __shared__ __align__(32) __bf16 sA[2][8][32][16];  // [buf][16-row blk][lane][elem]
  __shared__ __align__(32) __bf16 sB[2][8][32][16];  // [buf][16-col blk][lane][elem]
  const int tid  = threadIdx.x;
  const int lane = tid & 31;
  const int wv   = tid >> 5;          // wave 0..7; also the tile-record id
  const int wm   = wv >> 1;           // wave row 0..3  (32 rows each)
  const int wn   = wv & 1;            // wave col 0..1  (64 cols each)
  const int half = (lane >> 4) & 1;
  const int l15  = lane & 15;
  const int m0   = blockIdx.y * 128;
  const int n0   = blockIdx.x * 128;

  // Per-thread staging bases (record = [wv][lane]).
  const float* aRow = A + (size_t)(m0 + wv * 16 + l15) * K;        // + k0 + half*16
  const float* bRowT = transB ? (Bm + (size_t)(n0 + wv * 16 + l15) * K) : nullptr;
  const int    nColB = n0 + wv * 16 + l15;                          // transB=0 column

  auto stageA = [&](int buf, int k0) {
    const float* s = aRow + k0 + half * 16;
    float4 a0 = *(const float4*)(s + 0);
    float4 a1 = *(const float4*)(s + 4);
    float4 a2 = *(const float4*)(s + 8);
    float4 a3 = *(const float4*)(s + 12);
    unsigned* dst = (unsigned*)&sA[buf][wv][lane][0];
    ((uint4*)dst)[0] = make_uint4(pk2bf(a0.x, a0.y), pk2bf(a0.z, a0.w),
                                  pk2bf(a1.x, a1.y), pk2bf(a1.z, a1.w));
    ((uint4*)dst)[1] = make_uint4(pk2bf(a2.x, a2.y), pk2bf(a2.z, a2.w),
                                  pk2bf(a3.x, a3.y), pk2bf(a3.z, a3.w));
  };
  auto stageB = [&](int buf, int k0) {
    unsigned* dst = (unsigned*)&sB[buf][wv][lane][0];
    if (transB) {                                   // 16 contiguous floats along K
      const float* s = bRowT + k0 + half * 16;
      float4 b0 = *(const float4*)(s + 0);
      float4 b1 = *(const float4*)(s + 4);
      float4 b2 = *(const float4*)(s + 8);
      float4 b3 = *(const float4*)(s + 12);
      ((uint4*)dst)[0] = make_uint4(pk2bf(b0.x, b0.y), pk2bf(b0.z, b0.w),
                                    pk2bf(b1.x, b1.y), pk2bf(b1.z, b1.w));
      ((uint4*)dst)[1] = make_uint4(pk2bf(b2.x, b2.y), pk2bf(b2.z, b2.w),
                                    pk2bf(b3.x, b3.y), pk2bf(b3.z, b3.w));
    } else {                                        // N-strided quads along K
#pragma unroll
      for (int q = 0; q < 4; ++q) {
        const size_t base = (size_t)(k0 + half * 16 + q * 4) * N + nColB;
        const float v0 = Bm[base];
        const float v1 = Bm[base + (size_t)N];
        const float v2 = Bm[base + 2 * (size_t)N];
        const float v3 = Bm[base + 3 * (size_t)N];
        ((uint2*)dst)[q] = make_uint2(pk2bf(v0, v1), pk2bf(v2, v3));
      }
    }
  };

  v8f acc[2][4];
#pragma unroll
  for (int i = 0; i < 2; ++i)
#pragma unroll
    for (int j = 0; j < 4; ++j) acc[i][j] = (v8f){0.f,0.f,0.f,0.f,0.f,0.f,0.f,0.f};

  stageA(0, 0);
  stageB(0, 0);
  int buf = 0;
  for (int k0 = 0; k0 < K; k0 += 32) {
    __syncthreads();                 // staged tile [buf] visible to all waves
    if (k0 + 32 < K) {
      __builtin_prefetch(aRow + k0 + 64, 0, 1);   // global_prefetch_b8 ahead
      stageA(buf ^ 1, k0 + 32);
      stageB(buf ^ 1, k0 + 32);
    }
#pragma unroll
    for (int mi = 0; mi < 2; ++mi) {
      v16bf af = *(const v16bf*)(&sA[buf][wm * 2 + mi][lane][0]);
#pragma unroll
      for (int ni = 0; ni < 4; ++ni) {
        v16bf bf = *(const v16bf*)(&sB[buf][wn * 4 + ni][lane][0]);
        acc[mi][ni] = __builtin_amdgcn_wmma_f32_16x16x32_bf16(
            false, af, false, bf, (short)0, acc[mi][ni], false, false);
      }
    }
    buf ^= 1;
  }

  // --- epilogue: C layout lane 0-15 -> M=r, lane 16-31 -> M=r+8; N=lane&15 ---
#pragma unroll
  for (int mi = 0; mi < 2; ++mi)
#pragma unroll
    for (int ni = 0; ni < 4; ++ni) {
      const int col = n0 + (wn * 4 + ni) * 16 + l15;
      const float bv = (mode >= 1) ? bias[col] : 0.0f;
#pragma unroll
      for (int r = 0; r < 8; ++r) {
        const int row = m0 + (wm * 2 + mi) * 16 + half * 8 + r;
        float v = acc[mi][ni][r] + bv;
        if (mode == 2) v = 0.5f * v * (1.0f + erff(v * 0.70710678118f));
        if (mode == 3) v += Res[(size_t)row * N + col];
        C[(size_t)row * N + col] = v;
      }
    }
}

// ---------------------------------------------------------------------------
// Attention: one wave per (b,h,q).  ~5% of FLOPs -> f32 VALU.
// Q/K/V layout: [B, L, NH, DH] (natural reshape of GEMM output).
// ---------------------------------------------------------------------------
__global__ __launch_bounds__(256)
void attention_kernel(const float* __restrict__ Q, const float* __restrict__ K,
                      const float* __restrict__ V, float* __restrict__ O) {
  __shared__ float sc[8][SL];
  const int tid  = threadIdx.x;
  const int lane = tid & 31;
  const int wv   = tid >> 5;
  const int gw   = blockIdx.x * 8 + wv;            // 0 .. B*NH*L-1
  const int q = gw % SL;
  const int h = (gw / SL) % NHEAD;
  const int b = gw / (SL * NHEAD);
  const float scale = 0.125f;                      // 1/sqrt(64)

  const float* Qp = Q + (((size_t)(b * SL + q) * NHEAD + h) * DHEAD);
  float lmax = -3.4e38f;
  for (int k = lane; k < SL; k += 32) {
    const float* Kp = K + (((size_t)(b * SL + k) * NHEAD + h) * DHEAD);
    float s = 0.f;
#pragma unroll 8
    for (int d = 0; d < DHEAD; ++d) s += Qp[d] * Kp[d];
    s *= scale;
    sc[wv][k] = s;
    lmax = fmaxf(lmax, s);
  }
#pragma unroll
  for (int off = 16; off > 0; off >>= 1) lmax = fmaxf(lmax, __shfl_xor(lmax, off, 32));
  float lsum = 0.f;
  for (int k = lane; k < SL; k += 32) {
    float e = __expf(sc[wv][k] - lmax);
    sc[wv][k] = e; lsum += e;
  }
#pragma unroll
  for (int off = 16; off > 0; off >>= 1) lsum += __shfl_xor(lsum, off, 32);
  const float inv = 1.0f / lsum;
  __syncthreads();                                  // sc fully written per wave
  for (int d = lane; d < DHEAD; d += 32) {
    float o = 0.f;
    for (int k = 0; k < SL; ++k)
      o += sc[wv][k] * V[(((size_t)(b * SL + k) * NHEAD + h) * DHEAD) + d];
    O[(((size_t)(b * SL + q) * NHEAD + h) * DHEAD) + d] = o * inv;
  }
}

// ---------------------------------------------------------------------------
// LSTM direction sweep.  blockIdx.x = direction.  1024 threads; h/c in LDS.
// Gate precompute xW = X@Wih^T + bih already done by the WMMA GEMM.
// ---------------------------------------------------------------------------
__global__ __launch_bounds__(1024)
void lstm_dir_kernel(const float* __restrict__ xW,    // [2][MROWS][G4]
                     const float* __restrict__ Whh,   // [2][G4][HID]
                     const float* __restrict__ bhh,   // [2][G4]
                     float* __restrict__ out) {       // [B, L, 2*HID]
  __shared__ float hS[BSZ][HID];
  __shared__ float cS[BSZ][HID];
  const int tid = threadIdx.x;
  const int dir = blockIdx.x;
  const float* xw = xW  + (size_t)dir * MROWS * G4;
  const float* W  = Whh + (size_t)dir * G4 * HID;
  const float* bh = bhh + (size_t)dir * G4;
  for (int i = tid; i < BSZ * HID; i += 1024) { ((float*)hS)[i] = 0.f; ((float*)cS)[i] = 0.f; }
  __syncthreads();

  for (int s = 0; s < SL; ++s) {
    const int t = dir ? (SL - 1 - s) : s;
    float nh[3], nc[3];
#pragma unroll
    for (int u = 0; u < 3; ++u) {
      const int id = tid + u * 1024;       // 0..3071
      const int b  = id / HID;
      const int n  = id % HID;
      float gate[4];
#pragma unroll
      for (int gi = 0; gi < 4; ++gi) {
        const int j = gi * HID + n;
        float a = xw[((size_t)(b * SL + t)) * G4 + j] + bh[j];
        const float* wr = W + (size_t)j * HID;
        const float* hr = hS[b];
        float acc = 0.f;
        for (int k = 0; k < HID; ++k) acc += wr[k] * hr[k];
        gate[gi] = a + acc;
      }
      const float ig = sigmoidf_(gate[0]);
      const float fg = sigmoidf_(gate[1]);
      const float gg = tanhf(gate[2]);
      const float og = sigmoidf_(gate[3]);
      const float c  = fg * cS[b][n] + ig * gg;
      nc[u] = c;
      nh[u] = og * tanhf(c);
    }
    __syncthreads();
#pragma unroll
    for (int u = 0; u < 3; ++u) {
      const int id = tid + u * 1024;
      const int b  = id / HID;
      const int n  = id % HID;
      cS[b][n] = nc[u];
      hS[b][n] = nh[u];
      out[((size_t)(b * SL + t)) * (2 * HID) + dir * HID + n] = nh[u];
    }
    __syncthreads();
  }
}

// ---------------------------------------------------------------------------
// FC: feats[row, tag] = enc[row,:] . fc_w[tag,:] + fc_b[tag]
// ---------------------------------------------------------------------------
__global__ __launch_bounds__(256)
void fc_kernel(const float* __restrict__ X, const float* __restrict__ Wt,
               const float* __restrict__ bt, float* __restrict__ feats) {
  const int idx = blockIdx.x * 256 + threadIdx.x;
  if (idx >= MROWS * NTAG) return;
  const int row = idx / NTAG, tag = idx % NTAG;
  float acc = bt[tag];
  const float* xr = X + (size_t)row * DMODEL;
  const float* wr = Wt + (size_t)tag * DMODEL;
  for (int k = 0; k < DMODEL; ++k) acc += xr[k] * wr[k];
  feats[(size_t)row * NTAG + tag] = acc;
}

// ---------------------------------------------------------------------------
// Viterbi decode.  One block (32 thr) per batch element.
// out[b] = score; out[8 + b*L + t] = path tag (as float).
// ---------------------------------------------------------------------------
__global__ __launch_bounds__(32)
void viterbi_kernel(const float* __restrict__ feats, const float* __restrict__ trans,
                    int* __restrict__ bp, float* __restrict__ out) {
  __shared__ float fv[NTAG], nf[NTAG];
  const int b = blockIdx.x, j = threadIdx.x;
  if (j < NTAG) fv[j] = (j == TAG_CLS) ? 0.0f : NEGINF;
  __syncthreads();
  for (int t = 0; t < SL; ++t) {
    if (j < NTAG) {
      float best = -3.4e38f; int bi = 0;
      for (int i = 0; i < NTAG; ++i) {
        const float s = fv[i] + trans[j * NTAG + i];
        if (s > best) { best = s; bi = i; }
      }
      bp[((size_t)(b * SL + t)) * NTAG + j] = bi;
      nf[j] = best + feats[((size_t)(b * SL + t)) * NTAG + j];
    }
    __syncthreads();
    if (j < NTAG) fv[j] = nf[j];
    __syncthreads();
  }
  if (j == 0) {
    float best = -3.4e38f; int last = 0;
    for (int i = 0; i < NTAG; ++i) {
      const float s = fv[i] + trans[TAG_SEP * NTAG + i];
      if (s > best) { best = s; last = i; }
    }
    out[b] = best;
    int tag = last;
    for (int t = SL - 1; t >= 0; --t) {
      out[8 + (size_t)b * SL + t] = (float)tag;
      tag = bp[((size_t)(b * SL + t)) * NTAG + tag];
    }
  }
}

// ---------------------------------------------------------------------------
// Host-side orchestration (all launches on `stream`; workspace carved below).
// ---------------------------------------------------------------------------
extern "C" void kernel_launch(void* const* d_in, const int* in_sizes, int n_in,
                              void* d_out, int out_size, void* d_ws, size_t ws_size,
                              hipStream_t stream) {
  (void)in_sizes; (void)out_size; (void)ws_size;
  if (n_in < 29) return;
  const int*   sentence = (const int*)  d_in[0];
  const float* word_emb = (const float*)d_in[1];
  const float* pos_emb  = (const float*)d_in[2];
  const float* type_emb = (const float*)d_in[3];
  const float* emb_g    = (const float*)d_in[4];
  const float* emb_b    = (const float*)d_in[5];
  const float* Wq = (const float*)d_in[6];  const float* bq = (const float*)d_in[7];
  const float* Wk = (const float*)d_in[8];  const float* bk = (const float*)d_in[9];
  const float* Wv = (const float*)d_in[10]; const float* bv = (const float*)d_in[11];
  const float* Wo = (const float*)d_in[12]; const float* bo = (const float*)d_in[13];
  const float* ln1g = (const float*)d_in[14]; const float* ln1b = (const float*)d_in[15];
  const float* W1 = (const float*)d_in[16]; const float* b1 = (const float*)d_in[17];
  const float* W2 = (const float*)d_in[18]; const float* b2 = (const float*)d_in[19];
  const float* ln2g = (const float*)d_in[20]; const float* ln2b = (const float*)d_in[21];
  const float* Wih = (const float*)d_in[22]; const float* Whh = (const float*)d_in[23];
  const float* bih = (const float*)d_in[24]; const float* bhh = (const float*)d_in[25];
  const float* fc_w = (const float*)d_in[26]; const float* fc_b = (const float*)d_in[27];
  const float* trans = (const float*)d_in[28];

  // ---- workspace carve ----
  size_t off = 0;
  auto carve = [&](size_t bytes) -> char* {
    char* p = (char*)d_ws + off;
    off += (bytes + 255) & ~(size_t)255;
    return p;
  };
  float* X   = (float*)carve((size_t)MROWS * DMODEL * 4);  // activations
  float* T   = (float*)carve((size_t)MROWS * DMODEL * 4);  // pre-LN / LSTM out
  float* Qb  = (float*)carve((size_t)MROWS * DMODEL * 4);
  float* Kb  = (float*)carve((size_t)MROWS * DMODEL * 4);
  float* Vb  = (float*)carve((size_t)MROWS * DMODEL * 4);
  float* Ab  = (float*)carve((size_t)MROWS * DMODEL * 4);  // attention output
  float* Hbig= (float*)carve((size_t)MROWS * DFFN   * 4);  // FFN hidden / LSTM xW[2]
  float* feats=(float*)carve((size_t)MROWS * NTAG   * 4);
  int*   bp  = (int*)  carve((size_t)MROWS * NTAG   * 4);

  const dim3 blk256(256);
  auto gemm = [&](const float* A, const float* Bm, const float* bias,
                  const float* Res, float* C, int M, int N, int K,
                  int transB, int mode) {
    dim3 grid((N + 127) / 128, (M + 127) / 128);
    gemm_bf16_wmma<<<grid, blk256, 0, stream>>>(A, Bm, bias, Res, C, M, N, K, transB, mode);
  };

  // ---- embeddings + LN ----
  embed_ln_kernel<<<MROWS, blk256, 0, stream>>>(sentence, word_emb, pos_emb, type_emb,
                                                emb_g, emb_b, X);

  // ---- 12 BERT layers ----
  for (int l = 0; l < NLAY; ++l) {
    const size_t wo  = (size_t)l * DMODEL * DMODEL;
    const size_t bo1 = (size_t)l * DMODEL;
    gemm(X, Wq + wo, bq + bo1, nullptr, Qb, MROWS, DMODEL, DMODEL, 0, 1);
    gemm(X, Wk + wo, bk + bo1, nullptr, Kb, MROWS, DMODEL, DMODEL, 0, 1);
    gemm(X, Wv + wo, bv + bo1, nullptr, Vb, MROWS, DMODEL, DMODEL, 0, 1);
    attention_kernel<<<(BSZ * NHEAD * SL) / 8, blk256, 0, stream>>>(Qb, Kb, Vb, Ab);
    gemm(Ab, Wo + wo, bo + bo1, X, T, MROWS, DMODEL, DMODEL, 0, 3);            // +residual
    layernorm_kernel<<<MROWS, blk256, 0, stream>>>(T, ln1g + bo1, ln1b + bo1, X, DMODEL);
    gemm(X, W1 + (size_t)l * DMODEL * DFFN, b1 + (size_t)l * DFFN, nullptr,
         Hbig, MROWS, DFFN, DMODEL, 0, 2);                                     // +bias,GELU
    gemm(Hbig, W2 + (size_t)l * DFFN * DMODEL, b2 + bo1, X, T,
         MROWS, DMODEL, DFFN, 0, 3);                                           // +residual
    layernorm_kernel<<<MROWS, blk256, 0, stream>>>(T, ln2g + bo1, ln2b + bo1, X, DMODEL);
  }

  // ---- BiLSTM (2 layers x 2 dirs).  layer0: X -> T ; layer1: T -> X ----
  for (int lay = 0; lay < 2; ++lay) {
    const float* in  = (lay == 0) ? X : T;
    float*       outp = (lay == 0) ? T : X;
    for (int d = 0; d < 2; ++d) {
      const size_t wioff = ((size_t)(lay * 2 + d)) * G4 * DMODEL;
      const size_t boff  = ((size_t)(lay * 2 + d)) * G4;
      gemm(in, Wih + wioff, bih + boff, nullptr, Hbig + (size_t)d * MROWS * G4,
           MROWS, G4, DMODEL, 1, 1);   // x @ Wih^T + bih
    }
    lstm_dir_kernel<<<2, dim3(1024), 0, stream>>>(
        Hbig, Whh + (size_t)lay * 2 * G4 * HID, bhh + (size_t)lay * 2 * G4, outp);
  }

  // ---- FC + Viterbi ----
  fc_kernel<<<(MROWS * NTAG + 255) / 256, blk256, 0, stream>>>(X, fc_w, fc_b, feats);
  viterbi_kernel<<<BSZ, dim3(32), 0, stream>>>(feats, trans, bp, (float*)d_out);
}
